// GAT_GraphSVX_51402168598678
// MI455X (gfx1250) — matrix operand order repaired and checked by reading, not compiled, not run on hardware
//
#include <hip/hip_runtime.h>
#include <hip/hip_bf16.h>

typedef __attribute__((ext_vector_type(16))) __bf16 v16bf;
typedef __attribute__((ext_vector_type(8)))  float  v8f;

#define NNODES 50000
#define NEDGES 800000
#define EPLUS  850000   // edges + self loops
#define FDIM   128
#define NHEAD  4
#define HC     32
#define NEG_SLOPE 0.2f

// ---------------------------------------------------------------- utilities

__device__ __forceinline__ void atomicMaxF(float* addr, float val) {
    // works for mixed-sign floats given init = large negative
    if (val >= 0.0f) atomicMax((int*)addr, __float_as_int(val));
    else             atomicMin((unsigned int*)addr, __float_as_uint(val));
}

__device__ __forceinline__ void getEdge(int e, const int* __restrict__ ei,
                                        int& src, int& dst) {
    if (e < NEDGES) { src = ei[e]; dst = ei[NEDGES + e]; }
    else            { src = e - NEDGES; dst = src; }       // self loop
}

__device__ __forceinline__ float leaky(float a) {
    return a > 0.0f ? a : NEG_SLOPE * a;
}

// -------------------------------------------------- weight transpose+convert
// Wt[n*128 + k] = bf16( W[k*128 + n] )  -- puts each output column's K-run
// contiguous so the WMMA B fragment is two global_load_b128 per K-step.
__global__ void k_cvt_wt(const float* __restrict__ W, __bf16* __restrict__ Wt) {
    const int i = blockIdx.x * blockDim.x + threadIdx.x;   // n*128 + k
    if (i >= FDIM * FDIM) return;
    const int n = i >> 7, k = i & 127;
    Wt[i] = (__bf16)W[k * FDIM + n];
}

// ---------------------------------------------------------------- WMMA GEMM
// C[M,128] = A[M,128] @ W[128,128]; fp32 A in (cvt on the fly), bf16 Wt,
// f32 accumulate. Block = 256 threads (8 waves); block owns a 16x128 C tile,
// each wave one 16x16 tile, K=128 as 4 x v_wmma_f32_16x16x32_bf16.
__global__ __launch_bounds__(256) void k_gemm128(const float* __restrict__ A,
                                                 const __bf16* __restrict__ Wt,
                                                 float* __restrict__ C, int M) {
    const int wave = threadIdx.x >> 5;   // 0..7 -> output column tile
    const int lane = threadIdx.x & 31;
    const int half = lane >> 4;          // lane group (K split)
    const int l16  = lane & 15;
    const int row0 = blockIdx.x * 16;
    const int m    = row0 + l16;         // A-fragment row
    const int n    = wave * 16 + l16;    // B/C column
    if (row0 >= M) return;

    v8f acc = {};
#pragma unroll
    for (int kb = 0; kb < FDIM; kb += 32) {
        // A fragment: lanes 0-15 K=kb..kb+15, lanes 16-31 K=kb+16..kb+31 (matches B).
        // Per lane: one contiguous 16-float K-run -> vectorized b128 loads.
        v16bf a;
#pragma unroll
        for (int i = 0; i < 16; ++i) {
            const int k = 16 * half + i;
            a[i] = (__bf16)A[m * FDIM + kb + k];
        }
        // B fragment: lanes 0-15 K=kb..kb+15, lanes 16-31 K=kb+16..kb+31;
        // contiguous 32 bytes in Wt -> two global_load_b128.
        const v16bf b = *(const v16bf*)(Wt + n * FDIM + kb + 16 * half);
        acc = __builtin_amdgcn_wmma_f32_16x16x32_bf16(
            false, a, false, b, (short)0, acc, false, false);
    }
    // C/D layout: VGPR r -> row = r + 8*half, col = l16
#pragma unroll
    for (int r = 0; r < 8; ++r) {
        const int mr = row0 + r + 8 * half;
        if (mr < M) C[mr * FDIM + n] = acc[r];
    }
}

// ------------------------------------------------------- per-node alpha dots
__global__ void k_alpha(const float* __restrict__ h,
                        const float* __restrict__ att_src,
                        const float* __restrict__ att_dst,
                        float* __restrict__ aS, float* __restrict__ aD) {
    const int i = blockIdx.x * blockDim.x + threadIdx.x;   // node*H + head
    if (i >= NNODES * NHEAD) return;
    const int node = i >> 2, head = i & 3;
    const float* hp = h + node * FDIM + head * HC;
    const float* sp = att_src + head * HC;
    const float* dp = att_dst + head * HC;
    float s = 0.0f, d = 0.0f;
#pragma unroll
    for (int c = 0; c < HC; ++c) { const float v = hp[c]; s += v * sp[c]; d += v * dp[c]; }
    aS[i] = s; aD[i] = d;
}

// ------------------------------------------------------------ init buffers
__global__ void k_init(float* __restrict__ agg, float* __restrict__ amax,
                       float* __restrict__ denom) {
    const int i = blockIdx.x * blockDim.x + threadIdx.x;
    if (i < NNODES * FDIM) agg[i] = 0.0f;
    if (i < NNODES * NHEAD) { amax[i] = -3.0e38f; denom[i] = 0.0f; }
}

// ------------------------------------------------------------- edge kernels
__global__ void k_edge_max(const int* __restrict__ ei,
                           const float* __restrict__ aS, const float* __restrict__ aD,
                           float* __restrict__ amax) {
    const int t = blockIdx.x * blockDim.x + threadIdx.x;
    if (t >= EPLUS * NHEAD) return;
    const int e = t >> 2, head = t & 3;
    int src, dst; getEdge(e, ei, src, dst);
    const float a = leaky(aS[src * NHEAD + head] + aD[dst * NHEAD + head]);
    atomicMaxF(&amax[dst * NHEAD + head], a);
}

__global__ void k_edge_exp(const int* __restrict__ ei,
                           const float* __restrict__ aS, const float* __restrict__ aD,
                           const float* __restrict__ amax,
                           float* __restrict__ denom, float* __restrict__ eB) {
    const int t = blockIdx.x * blockDim.x + threadIdx.x;
    if (t >= EPLUS * NHEAD) return;
    const int e = t >> 2, head = t & 3;
    int src, dst; getEdge(e, ei, src, dst);
    const float a  = leaky(aS[src * NHEAD + head] + aD[dst * NHEAD + head]);
    const float ex = __expf(a - amax[dst * NHEAD + head]);
    eB[t] = ex;
    atomicAdd(&denom[dst * NHEAD + head], ex);
}

// one thread per (edge, 4 channels): b128 gather of h[src], 4 f32 atomics
__global__ void k_edge_agg(const int* __restrict__ ei,
                           const float* __restrict__ h,
                           const float* __restrict__ eB,
                           const float* __restrict__ denom,
                           float* __restrict__ agg) {
    const int t = blockIdx.x * blockDim.x + threadIdx.x;   // edge*32 + c4
    if (t >= EPLUS * (FDIM / 4)) return;
    const int e = t >> 5, c = (t & 31) * 4, head = c >> 5;
    int src, dst; getEdge(e, ei, src, dst);
    const float w = eB[e * NHEAD + head] / denom[dst * NHEAD + head];
    const float4 hv = *(const float4*)(h + src * FDIM + c);
    float* ap = agg + dst * FDIM + c;
    atomicAdd(ap + 0, hv.x * w);
    atomicAdd(ap + 1, hv.y * w);
    atomicAdd(ap + 2, hv.z * w);
    atomicAdd(ap + 3, hv.w * w);
}

__global__ void k_bias_elu(float* __restrict__ buf, const float* __restrict__ b) {
    const int i = blockIdx.x * blockDim.x + threadIdx.x;
    if (i >= NNODES * FDIM) return;
    const float v = buf[i] + b[i & 127];
    buf[i] = v > 0.0f ? v : (__expf(v) - 1.0f);
}

// --------------------------------------------------------------- layer 3
__global__ void k_l3_gemm(const float* __restrict__ h2, const float* __restrict__ W3,
                          const float* __restrict__ as3w, const float* __restrict__ ad3w,
                          float* __restrict__ h3, float* __restrict__ aS3,
                          float* __restrict__ aD3) {
    const int i = blockIdx.x * blockDim.x + threadIdx.x;
    if (i >= NNODES) return;
    float o0 = 0.0f, o1 = 0.0f;
    const float* hp = h2 + i * FDIM;
#pragma unroll 8
    for (int k = 0; k < FDIM; ++k) {
        const float v = hp[k];
        o0 += v * W3[k * 2];
        o1 += v * W3[k * 2 + 1];
    }
    h3[i * 2] = o0; h3[i * 2 + 1] = o1;
    aS3[i] = o0 * as3w[0] + o1 * as3w[1];
    aD3[i] = o0 * ad3w[0] + o1 * ad3w[1];
}

__global__ void k_init3(float* __restrict__ agg3, float* __restrict__ amax3,
                        float* __restrict__ denom3) {
    const int i = blockIdx.x * blockDim.x + threadIdx.x;
    if (i >= NNODES) return;
    amax3[i] = -3.0e38f; denom3[i] = 0.0f;
    agg3[i * 2] = 0.0f; agg3[i * 2 + 1] = 0.0f;
}

__global__ void k_edge_max3(const int* __restrict__ ei, const float* __restrict__ aS3,
                            const float* __restrict__ aD3, float* __restrict__ amax3) {
    const int e = blockIdx.x * blockDim.x + threadIdx.x;
    if (e >= EPLUS) return;
    int src, dst; getEdge(e, ei, src, dst);
    atomicMaxF(&amax3[dst], leaky(aS3[src] + aD3[dst]));
}

__global__ void k_edge_exp3(const int* __restrict__ ei, const float* __restrict__ aS3,
                            const float* __restrict__ aD3, const float* __restrict__ amax3,
                            float* __restrict__ denom3, float* __restrict__ e3) {
    const int e = blockIdx.x * blockDim.x + threadIdx.x;
    if (e >= EPLUS) return;
    int src, dst; getEdge(e, ei, src, dst);
    const float ex = __expf(leaky(aS3[src] + aD3[dst]) - amax3[dst]);
    e3[e] = ex;
    atomicAdd(&denom3[dst], ex);
}

__global__ void k_edge_agg3(const int* __restrict__ ei, const float* __restrict__ h3,
                            const float* __restrict__ e3, const float* __restrict__ denom3,
                            float* __restrict__ agg3) {
    const int e = blockIdx.x * blockDim.x + threadIdx.x;
    if (e >= EPLUS) return;
    int src, dst; getEdge(e, ei, src, dst);
    const float w = e3[e] / denom3[dst];
    atomicAdd(&agg3[dst * 2],     h3[src * 2]     * w);
    atomicAdd(&agg3[dst * 2 + 1], h3[src * 2 + 1] * w);
}

__global__ void k_final(const float* __restrict__ agg3, const float* __restrict__ b3,
                        float* __restrict__ out) {
    const int i = blockIdx.x * blockDim.x + threadIdx.x;
    if (i >= NNODES) return;
    const float v0 = agg3[i * 2]     + b3[0];
    const float v1 = agg3[i * 2 + 1] + b3[1];
    const float m  = fmaxf(v0, v1);
    const float lse = m + __logf(__expf(v0 - m) + __expf(v1 - m));
    out[i * 2]     = v0 - lse;
    out[i * 2 + 1] = v1 - lse;
}

// ---------------------------------------------------------------- launcher
extern "C" void kernel_launch(void* const* d_in, const int* in_sizes, int n_in,
                              void* d_out, int out_size, void* d_ws, size_t ws_size,
                              hipStream_t stream) {
    const float* x   = (const float*)d_in[0];
    const int*   ei  = (const int*)  d_in[1];
    const float* W1  = (const float*)d_in[2];
    const float* as1 = (const float*)d_in[3];
    const float* ad1 = (const float*)d_in[4];
    const float* b1  = (const float*)d_in[5];
    const float* W2  = (const float*)d_in[6];
    const float* as2 = (const float*)d_in[7];
    const float* ad2 = (const float*)d_in[8];
    const float* b2  = (const float*)d_in[9];
    const float* W3  = (const float*)d_in[10];
    const float* as3w= (const float*)d_in[11];
    const float* ad3w= (const float*)d_in[12];
    const float* b3  = (const float*)d_in[13];

    // workspace layout (floats)
    float* ws = (float*)d_ws;
    float* P    = ws;                      // [N,128] gemm output
    float* Q    = P    + NNODES * FDIM;    // [N,128] aggregate / activations
    float* aS   = Q    + NNODES * FDIM;    // [N,H]
    float* aD   = aS   + NNODES * NHEAD;
    float* aM   = aD   + NNODES * NHEAD;
    float* dn   = aM   + NNODES * NHEAD;
    float* eB   = dn   + NNODES * NHEAD;   // [E',H]
    float* h3   = eB   + EPLUS * NHEAD;    // [N,2]
    float* aS3  = h3   + NNODES * 2;
    float* aD3  = aS3  + NNODES;
    float* aM3  = aD3  + NNODES;
    float* dn3  = aM3  + NNODES;
    float* e3   = dn3  + NNODES;           // [E']
    float* g3   = e3   + EPLUS;            // [N,2]
    __bf16* Wt  = (__bf16*)(g3 + NNODES * 2);  // [128,128] bf16, 32B-aligned

    const int T = 256;
    const int gNode   = (NNODES + T - 1) / T;
    const int gNH     = (NNODES * NHEAD + T - 1) / T;
    const int gNF     = (NNODES * FDIM + T - 1) / T;
    const int gEH     = (EPLUS * NHEAD + T - 1) / T;
    const int gEC     = (EPLUS * (FDIM / 4) + T - 1) / T;
    const int gE      = (EPLUS + T - 1) / T;
    const int gGemm   = (NNODES + 15) / 16;
    const int gW      = (FDIM * FDIM + T - 1) / T;

    // ---------------- layer 1 : x -> Q (elu'd h1)
    k_cvt_wt  <<<gW,    T, 0, stream>>>(W1, Wt);
    k_gemm128 <<<gGemm, T, 0, stream>>>(x, Wt, P, NNODES);
    k_alpha   <<<gNH,   T, 0, stream>>>(P, as1, ad1, aS, aD);
    k_init    <<<gNF,   T, 0, stream>>>(Q, aM, dn);
    k_edge_max<<<gEH,   T, 0, stream>>>(ei, aS, aD, aM);
    k_edge_exp<<<gEH,   T, 0, stream>>>(ei, aS, aD, aM, dn, eB);
    k_edge_agg<<<gEC,   T, 0, stream>>>(ei, P, eB, dn, Q);
    k_bias_elu<<<gNF,   T, 0, stream>>>(Q, b1);

    // ---------------- layer 2 : Q -> Q (elu'd h2)
    k_cvt_wt  <<<gW,    T, 0, stream>>>(W2, Wt);
    k_gemm128 <<<gGemm, T, 0, stream>>>(Q, Wt, P, NNODES);
    k_alpha   <<<gNH,   T, 0, stream>>>(P, as2, ad2, aS, aD);
    k_init    <<<gNF,   T, 0, stream>>>(Q, aM, dn);
    k_edge_max<<<gEH,   T, 0, stream>>>(ei, aS, aD, aM);
    k_edge_exp<<<gEH,   T, 0, stream>>>(ei, aS, aD, aM, dn, eB);
    k_edge_agg<<<gEC,   T, 0, stream>>>(ei, P, eB, dn, Q);
    k_bias_elu<<<gNF,   T, 0, stream>>>(Q, b2);

    // ---------------- layer 3 : Q -> d_out (log_softmax)
    k_l3_gemm  <<<gNode, T, 0, stream>>>(Q, W3, as3w, ad3w, h3, aS3, aD3);
    k_init3    <<<gNode, T, 0, stream>>>(g3, aM3, dn3);
    k_edge_max3<<<gE,    T, 0, stream>>>(ei, aS3, aD3, aM3);
    k_edge_exp3<<<gE,    T, 0, stream>>>(ei, aS3, aD3, aM3, dn3, e3);
    k_edge_agg3<<<gE,    T, 0, stream>>>(ei, h3, e3, dn3, g3);
    k_final    <<<gNode, T, 0, stream>>>(g3, b3, (float*)d_out);
}
